// SelfAttention_858993459582
// MI455X (gfx1250) — matrix operand order repaired and checked by reading, not compiled, not run on hardware
//
#include <hip/hip_runtime.h>

// ---------------------------------------------------------------------------
// Types for CDNA5 WMMA (wave32)
// ---------------------------------------------------------------------------
typedef __bf16 bf16;
typedef __attribute__((ext_vector_type(16))) __bf16 v16bf;
typedef __attribute__((ext_vector_type(8)))  __bf16 v8bf;
typedef __attribute__((ext_vector_type(8)))  float  v8f;

#define DEVINL __device__ __forceinline__

// ---------------------------------------------------------------------------
// Problem constants
//   in_data  : [128][256][256] f32 (B1=1)
//   mask     : [128][256] f32
//   nb_bias  : [8][256][256] f32
//   w_qkv    : [768][256] f32, w_gate : [256][256], w_o : [256][256]
// ---------------------------------------------------------------------------
#define NTOK   32768   // 128*256 tokens
#define DMODEL 256
#define NHEAD  8
#define HDIM   32
#define SEQ    256
#define NB2    128

// ---------------------------------------------------------------------------
// Fragment loaders (bf16 16x16x32 WMMA).
// A (16xK=32): lane l -> row M=(l&15); per-lane 16 values:
//   elems 0..7  = K[kg*8 .. kg*8+7],  elems 8..15 = K[16+kg*8 .. 16+kg*8+7]
//   where kg = l>>4.  (ISA 7.12.2, 16-bit A 16x32 table)
// B (K=32 x 16): lane l -> col N=(l&15); per-lane 16 values in same K order as A:
//   elems 0..7 = K[kg*8 .. kg*8+7], elems 8..15 = K[16+kg*8 .. 16+kg*8+7]
// C/D (16x16 f32): VGPR r -> row M=r (+8 for lanes 16..31), lane&15 -> N.
// ---------------------------------------------------------------------------
DEVINL v16bf load_a_frag(const bf16* __restrict__ src, int ld, int row0, int k0, int lane) {
    const int r  = row0 + (lane & 15);
    const int kb = k0 + ((lane >> 4) << 3);
    const bf16* p = src + (size_t)r * (size_t)ld + kb;
    v8bf lo = *(const v8bf*)(p);
    v8bf hi = *(const v8bf*)(p + 16);
    return __builtin_shufflevector(lo, hi, 0,1,2,3,4,5,6,7,8,9,10,11,12,13,14,15);
}

DEVINL v16bf load_b_frag(const bf16* __restrict__ src, int ld, int n0, int k0, int lane) {
    const int n  = n0 + (lane & 15);
    const int kb = k0 + ((lane >> 4) << 3);
    const bf16* p = src + (size_t)n * (size_t)ld + kb;
    v8bf lo = *(const v8bf*)(p);
    v8bf hi = *(const v8bf*)(p + 16);
    return __builtin_shufflevector(lo, hi, 0,1,2,3,4,5,6,7,8,9,10,11,12,13,14,15);
}

DEVINL v8f wmma_bf16(v16bf a, v16bf b, v8f c) {
    return __builtin_amdgcn_wmma_f32_16x16x32_bf16(
        /*neg_a=*/false, a, /*neg_b=*/false, b,
        /*c_mod=*/(short)0, c, /*reuse_a=*/false, /*reuse_b=*/false);
}

DEVINL v8f vzero8() { v8f z = {0.f,0.f,0.f,0.f,0.f,0.f,0.f,0.f}; return z; }

// ---------------------------------------------------------------------------
// Kernel 1: fp32 -> bf16 conversion of activations and weights.
//   X  [32768][256]  <- in_data
//   W  [1024][256]   <- rows 0..767 w_qkv, 768..1023 w_gate
//   Wo [256][256]    <- w_o
// ---------------------------------------------------------------------------
#define CONV_X   8388608u               // 32768*256
#define CONV_WQ  196608u                // 768*256
#define CONV_WG  65536u                 // 256*256
#define CONV_WO  65536u
#define CONV_TOT (CONV_X + CONV_WQ + CONV_WG + CONV_WO)   // 8716288

__global__ __launch_bounds__(256)
void convert_kernel(const float* __restrict__ in_data,
                    const float* __restrict__ w_qkv,
                    const float* __restrict__ w_gate,
                    const float* __restrict__ w_o,
                    bf16* __restrict__ X, bf16* __restrict__ W, bf16* __restrict__ Wo) {
    unsigned i = blockIdx.x * 256u + threadIdx.x;
    if (i < CONV_X) {
        X[i] = (bf16)in_data[i];
    } else if (i < CONV_X + CONV_WQ) {
        unsigned j = i - CONV_X;
        W[j] = (bf16)w_qkv[j];
    } else if (i < CONV_X + CONV_WQ + CONV_WG) {
        unsigned j = i - (CONV_X + CONV_WQ);
        W[CONV_WQ + j] = (bf16)w_gate[j];
    } else if (i < CONV_TOT) {
        unsigned j = i - (CONV_X + CONV_WQ + CONV_WG);
        Wo[j] = (bf16)w_o[j];
    }
}

// ---------------------------------------------------------------------------
// Kernel 2: fused QKV+gate projection. GEMM [32768 x 256] x [256 x 1024].
// One wave -> 16(M) x 64(N) tile; 8 waves per block; 4096 blocks.
// Epilogue by feature section:
//   e in [0,256)    q : *C^-0.5, store [b2][h][n][32]
//   e in [256,512)  k : store [b2][h][n][32]
//   e in [512,768)  v : store TRANSPOSED [b2][h][32][n]
//   e in [768,1024) gate : sigmoid(x + gating_bias) fp32 [tok][256]
// ---------------------------------------------------------------------------
__global__ __launch_bounds__(256)
void proj_qkvg_kernel(const bf16* __restrict__ X, const bf16* __restrict__ W,
                      const float* __restrict__ gating_bias,
                      bf16* __restrict__ q, bf16* __restrict__ k,
                      bf16* __restrict__ vT, float* __restrict__ gate) {
    const int lane = threadIdx.x & 31;
    const int wid  = threadIdx.x >> 5;
    const int gw   = blockIdx.x * 8 + wid;   // 0..32767
    const int mt   = gw >> 4;                // 0..2047
    const int ng   = gw & 15;                // 0..15
    const int row0 = mt << 4;
    const int n0   = ng << 6;

    v8f acc[4];
#pragma unroll
    for (int j = 0; j < 4; ++j) acc[j] = vzero8();

#pragma unroll
    for (int kt = 0; kt < 8; ++kt) {
        v16bf a = load_a_frag(X, DMODEL, row0, kt * 32, lane);
#pragma unroll
        for (int j = 0; j < 4; ++j) {
            v16bf b = load_b_frag(W, DMODEL, n0 + j * 16, kt * 32, lane);
            acc[j] = wmma_bf16(a, b, acc[j]);
        }
    }

    const int lo = lane & 15;
    const int hi = lane >> 4;
#pragma unroll
    for (int j = 0; j < 4; ++j) {
        const int e0  = n0 + j * 16;
        const int sec = e0 >> 8;          // uniform per wave (n0 % 64 == 0)
        const int e   = (e0 & 255) + lo;  // feature within section
        const int h   = e >> 5;
        const int c   = e & 31;
#pragma unroll
        for (int r = 0; r < 8; ++r) {
            const int tok = row0 + r + hi * 8;
            const int b2  = tok >> 8;
            const int n   = tok & 255;
            const float val = acc[j][r];
            if (sec == 0) {
                q[((size_t)((b2 * NHEAD + h) * SEQ + n)) * HDIM + c] =
                    (bf16)(val * 0.17677669529663687f);   // 32^-0.5
            } else if (sec == 1) {
                k[((size_t)((b2 * NHEAD + h) * SEQ + n)) * HDIM + c] = (bf16)val;
            } else if (sec == 2) {
                vT[((size_t)((b2 * NHEAD + h) * HDIM + c)) * SEQ + n] = (bf16)val;
            } else {
                const float g = val + gating_bias[e];
                gate[(size_t)tok * DMODEL + e] = 1.0f / (1.0f + __expf(-g));
            }
        }
    }
}

// ---------------------------------------------------------------------------
// Kernel 3: attention core. One block per (b2, head), 128 threads = 4 waves.
// Each wave handles 4 row-tiles of 16 query rows sequentially:
//   logits  : 16 WMMAs (full K=C=32 in one), 16x256 f32 kept in 128 VGPRs
//   softmax : row reductions via 16-lane shfl_xor (C rows span lanes 0..15)
//   p       : re-striped C-layout -> A-layout through LDS (bf16)
//   out     : 2x8 WMMAs against vT, gated, stored bf16 [tok][256]
// ---------------------------------------------------------------------------
__global__ __launch_bounds__(128)
void attn_kernel(const bf16* __restrict__ q, const bf16* __restrict__ k,
                 const bf16* __restrict__ vT,
                 const float* __restrict__ mask,
                 const float* __restrict__ nb_bias,
                 const float* __restrict__ gate,
                 bf16* __restrict__ gated) {
    __shared__ bf16 sP[4][16][SEQ];   // 32 KB: per-wave p staging buffer

    const int lane = threadIdx.x & 31;
    const int wid  = threadIdx.x >> 5;
    const int b2   = blockIdx.x >> 3;
    const int h    = blockIdx.x & 7;
    const int lo   = lane & 15;
    const int hi   = lane >> 4;

    const bf16*  qh    = q  + (size_t)(b2 * NHEAD + h) * SEQ * HDIM;
    const bf16*  kh    = k  + (size_t)(b2 * NHEAD + h) * SEQ * HDIM;
    const bf16*  vh    = vT + (size_t)(b2 * NHEAD + h) * HDIM * SEQ;
    const float* maskb = mask + (size_t)b2 * SEQ;
    const float* biash = nb_bias + (size_t)h * SEQ * SEQ;

    for (int it = 0; it < 4; ++it) {
        const int rt   = wid + it * 4;   // row tile 0..15
        const int row0 = rt * 16;

        // ---- logits = (q*scale) @ k^T : one A-frag, 16 key tiles ----
        v16bf aq = load_a_frag(qh, HDIM, row0, 0, lane);
        v8f acc[16];
#pragma unroll
        for (int kt = 0; kt < 16; ++kt) {
            v16bf bk = load_b_frag(kh, HDIM, kt * 16, 0, lane);
            acc[kt] = wmma_bf16(aq, bk, vzero8());
        }

        // ---- + mask bias + nonbatched bias ----
#pragma unroll
        for (int kt = 0; kt < 16; ++kt) {
            const int key = kt * 16 + lo;
            const float mb = (maskb[key] - 1.0f) * 1e9f;
#pragma unroll
            for (int r = 0; r < 8; ++r) {
                const int rowg = row0 + r + hi * 8;
                acc[kt][r] = acc[kt][r] + mb + biash[rowg * SEQ + key];
            }
        }

        // ---- row softmax (rows live across lanes {0..15} / {16..31}) ----
#pragma unroll
        for (int r = 0; r < 8; ++r) {
            float m = -3.0e38f;
#pragma unroll
            for (int kt = 0; kt < 16; ++kt) m = fmaxf(m, acc[kt][r]);
            m = fmaxf(m, __shfl_xor(m, 1, 32));
            m = fmaxf(m, __shfl_xor(m, 2, 32));
            m = fmaxf(m, __shfl_xor(m, 4, 32));
            m = fmaxf(m, __shfl_xor(m, 8, 32));
            float s = 0.f;
#pragma unroll
            for (int kt = 0; kt < 16; ++kt) {
                const float e = __expf(acc[kt][r] - m);
                acc[kt][r] = e;
                s += e;
            }
            s += __shfl_xor(s, 1, 32);
            s += __shfl_xor(s, 2, 32);
            s += __shfl_xor(s, 4, 32);
            s += __shfl_xor(s, 8, 32);
            const float inv = 1.0f / s;
#pragma unroll
            for (int kt = 0; kt < 16; ++kt) acc[kt][r] = acc[kt][r] * inv;
        }

        // ---- re-stripe p through LDS: C-layout -> A-layout ----
#pragma unroll
        for (int kt = 0; kt < 16; ++kt)
#pragma unroll
            for (int r = 0; r < 8; ++r)
                sP[wid][r + hi * 8][kt * 16 + lo] = (bf16)acc[kt][r];

        // ---- weighted avg: p[16x256] @ v[256x32] ----
#pragma unroll
        for (int ct = 0; ct < 2; ++ct) {
            v8f o = vzero8();
#pragma unroll
            for (int k8 = 0; k8 < 8; ++k8) {
                v16bf ap = load_a_frag(&sP[wid][0][0], SEQ, 0, k8 * 32, lane);
                v16bf bv = load_b_frag(vh, SEQ, ct * 16, k8 * 32, lane);
                o = wmma_bf16(ap, bv, o);
            }
            // ---- gate + store ----
#pragma unroll
            for (int r = 0; r < 8; ++r) {
                const int rowg = row0 + r + hi * 8;
                const int c    = ct * 16 + lo;
                const size_t idx = (size_t)(b2 * SEQ + rowg) * DMODEL + h * HDIM + c;
                gated[idx] = (bf16)(o[r] * gate[idx]);
            }
        }
    }
}

// ---------------------------------------------------------------------------
// Kernel 4: output projection. GEMM [32768 x 256] x [256 x 256] + b_o -> f32.
// One wave -> 16(M) x 64(N); 8 waves/block; 1024 blocks.
// ---------------------------------------------------------------------------
__global__ __launch_bounds__(256)
void out_proj_kernel(const bf16* __restrict__ G, const bf16* __restrict__ Wo,
                     const float* __restrict__ b_o, float* __restrict__ out) {
    const int lane = threadIdx.x & 31;
    const int wid  = threadIdx.x >> 5;
    const int gw   = blockIdx.x * 8 + wid;   // 0..8191
    const int mt   = gw >> 2;
    const int ng   = gw & 3;
    const int row0 = mt << 4;
    const int n0   = ng << 6;

    v8f acc[4];
#pragma unroll
    for (int j = 0; j < 4; ++j) acc[j] = vzero8();

#pragma unroll
    for (int kt = 0; kt < 8; ++kt) {
        v16bf a = load_a_frag(G, DMODEL, row0, kt * 32, lane);
#pragma unroll
        for (int j = 0; j < 4; ++j) {
            v16bf b = load_b_frag(Wo, DMODEL, n0 + j * 16, kt * 32, lane);
            acc[j] = wmma_bf16(a, b, acc[j]);
        }
    }

    const int lo = lane & 15;
    const int hi = lane >> 4;
#pragma unroll
    for (int j = 0; j < 4; ++j) {
        const int o = n0 + j * 16 + lo;
        const float bias = b_o[o];
#pragma unroll
        for (int r = 0; r < 8; ++r) {
            const int tok = row0 + r + hi * 8;
            out[(size_t)tok * DMODEL + o] = acc[j][r] + bias;
        }
    }
}

// ---------------------------------------------------------------------------
// Workspace layout (bytes, 256-aligned):
//   X     @ 0           16,777,216   bf16 [32768][256]
//   W     @ 16777216       524,288   bf16 [1024][256]
//   Wo    @ 17301504       131,072   bf16 [256][256]
//   q     @ 17432576    16,777,216   bf16 [128][8][256][32]
//   k     @ 34209792    16,777,216   bf16 [128][8][256][32]
//   vT    @ 50987008    16,777,216   bf16 [128][8][32][256]
//   gate  @ 67764224    33,554,432   f32  [32768][256]
//   G     @ 101318656   16,777,216   bf16 [32768][256]
//   total ~ 118.1 MB
// ---------------------------------------------------------------------------
extern "C" void kernel_launch(void* const* d_in, const int* in_sizes, int n_in,
                              void* d_out, int out_size, void* d_ws, size_t ws_size,
                              hipStream_t stream) {
    (void)in_sizes; (void)n_in; (void)out_size; (void)ws_size;

    const float* in_data     = (const float*)d_in[0];
    const float* mask        = (const float*)d_in[1];
    const float* nb_bias     = (const float*)d_in[2];
    const float* w_qkv       = (const float*)d_in[3];
    const float* w_gate      = (const float*)d_in[4];
    const float* gating_bias = (const float*)d_in[5];
    const float* w_o         = (const float*)d_in[6];
    const float* b_o         = (const float*)d_in[7];
    float*       out         = (float*)d_out;

    char* ws = (char*)d_ws;
    bf16*  X    = (bf16*)(ws + 0);
    bf16*  W    = (bf16*)(ws + 16777216);
    bf16*  Wo   = (bf16*)(ws + 17301504);
    bf16*  q    = (bf16*)(ws + 17432576);
    bf16*  k    = (bf16*)(ws + 34209792);
    bf16*  vT   = (bf16*)(ws + 50987008);
    float* gate = (float*)(ws + 67764224);
    bf16*  G    = (bf16*)(ws + 101318656);

    convert_kernel<<<(CONV_TOT + 255u) / 256u, 256, 0, stream>>>(
        in_data, w_qkv, w_gate, w_o, X, W, Wo);

    proj_qkvg_kernel<<<4096, 256, 0, stream>>>(X, W, gating_bias, q, k, vT, gate);

    attn_kernel<<<NB2 * NHEAD, 128, 0, stream>>>(q, k, vT, mask, nb_bias, gate, G);

    out_proj_kernel<<<1024, 256, 0, stream>>>(G, Wo, b_o, out);
}
